// MultiHeadAttention_46024869544096
// MI455X (gfx1250) — compile-verified, matrix operand order checked by reading, not run on hardware
//
#include <hip/hip_runtime.h>
#include <hip/hip_bf16.h>

// MHA forward for MI455X (gfx1250, wave32, WMMA), fused-attention version.
// d_out = [heads (B*S*D f32) | alignment (B*H*S*S f32)]
// d_ws  = Qp | Kp | Vp | ctx, each B*S*D f16 (32 MB total)

typedef __attribute__((ext_vector_type(16))) _Float16 v16h;
typedef __attribute__((ext_vector_type(8)))  _Float16 v8h;
typedef __attribute__((ext_vector_type(8)))  float    v8f;

constexpr int Bc = 2;
constexpr int Sc = 2048;
constexpr int Dc = 1024;
constexpr int Hc = 16;
constexpr int Kc = 64;           // head dim
constexpr int LSTR = 40;         // GEMM LDS stride (halfs): 80B, 16B-aligned, skewed
constexpr int QSTR = 72;         // Q/K tile stride (64 k + pad): 144B, 16B-aligned
constexpr int PSTR = 136;        // P/V tile stride (128 k + pad): 272B, 16B-aligned

__device__ __forceinline__ v8f wmma16x16x32(v16h a, v16h b, v8f c) {
  return __builtin_amdgcn_wmma_f32_16x16x32_f16(false, a, false, b, (short)0, c,
                                                false, false);
}

// 16x32 f16 A/B fragment from LDS tile [rows][stride], k-window at kb.
// Lanes 0-15: row M=lane, K={kb+0..7, kb+16..23}; lanes 16-31: K={kb+8..15, kb+24..31}.
__device__ __forceinline__ v16h load_frag_s(const _Float16* tile, int stride,
                                            int row0, int kb, int lane) {
  const int hsel = lane >> 4;
  const int lr   = lane & 15;
  const _Float16* p = tile + (row0 + lr) * stride + kb;
  const int k0 = hsel ? 8 : 0;
  const int k1 = hsel ? 24 : 16;
  union { v16h v; v8h h[2]; } u;
  u.h[0] = *(const v8h*)(p + k0);
  u.h[1] = *(const v8h*)(p + k1);
  return u.v;
}

// ---------------------------------------------------------------------------
// Generic GEMM + bias: out[M,N] = A[M,K] @ W[K,N] + bias[N]
// ---------------------------------------------------------------------------
template <typename TA, typename TO>
__global__ void __launch_bounds__(256)
gemm_bias_kernel(const TA* __restrict__ A, const float* __restrict__ W,
                 const float* __restrict__ bias, TO* __restrict__ out,
                 int M, int N, int K) {
  __shared__ _Float16 As[128 * LSTR];       // [m][k]
  __shared__ _Float16 Bs[128 * LSTR];       // [n][k]

  const int t    = threadIdx.x;
  const int lane = t & 31;
  const int wid  = t >> 5;
  const int m0   = blockIdx.x * 128;
  const int n0   = blockIdx.y * 128;
  const int wm   = (wid >> 1) * 32;
  const int wn   = (wid & 1) * 64;

  const int ar  = t >> 1;
  const int ac  = (t & 1) * 16;
  const int bkr = t >> 3;
  const int bnc = (t & 7) * 16;

  v8f acc[2][4] = {};

  for (int kb = 0; kb < K; kb += 32) {
    {
      const TA* src = A + (size_t)(m0 + ar) * K + kb + ac;
      _Float16* dst = As + ar * LSTR + ac;
      if (sizeof(TA) == 4) {
        const float4* s4 = (const float4*)src;
#pragma unroll
        for (int j = 0; j < 4; ++j) {
          float4 vv = s4[j];
          dst[j * 4 + 0] = (_Float16)vv.x;
          dst[j * 4 + 1] = (_Float16)vv.y;
          dst[j * 4 + 2] = (_Float16)vv.z;
          dst[j * 4 + 3] = (_Float16)vv.w;
        }
      } else {
        const v8h* s8 = (const v8h*)src;
        *(v8h*)(dst)     = s8[0];
        *(v8h*)(dst + 8) = s8[1];
      }
      if (kb + 32 < K)
        __builtin_prefetch(A + (size_t)(m0 + ar) * K + kb + 32 + ac, 0, 1);
    }
    {
      const float4* s4 = (const float4*)(W + (size_t)(kb + bkr) * N + n0 + bnc);
#pragma unroll
      for (int j4 = 0; j4 < 4; ++j4) {
        float4 vv = s4[j4];
        _Float16* d = Bs + (bnc + j4 * 4) * LSTR + bkr;
        d[0 * LSTR] = (_Float16)vv.x;
        d[1 * LSTR] = (_Float16)vv.y;
        d[2 * LSTR] = (_Float16)vv.z;
        d[3 * LSTR] = (_Float16)vv.w;
      }
      if (kb + 32 < K)
        __builtin_prefetch(W + (size_t)(kb + 32 + bkr) * N + n0 + bnc, 0, 1);
    }
    __syncthreads();

    v16h af[2], bf[4];
#pragma unroll
    for (int i = 0; i < 2; ++i) af[i] = load_frag_s(As, LSTR, wm + i * 16, 0, lane);
#pragma unroll
    for (int j = 0; j < 4; ++j) bf[j] = load_frag_s(Bs, LSTR, wn + j * 16, 0, lane);
#pragma unroll
    for (int i = 0; i < 2; ++i)
#pragma unroll
      for (int j = 0; j < 4; ++j)
        acc[i][j] = wmma16x16x32(af[i], bf[j], acc[i][j]);
    __syncthreads();
  }

  const int hsel = lane >> 4, lr = lane & 15;
#pragma unroll
  for (int i = 0; i < 2; ++i) {
#pragma unroll
    for (int j = 0; j < 4; ++j) {
      const int gn = n0 + wn + j * 16 + lr;
      const float bv = bias[gn];
#pragma unroll
      for (int r = 0; r < 8; ++r) {
        const int gm = m0 + wm + i * 16 + hsel * 8 + r;
        out[(size_t)gm * N + gn] = (TO)(acc[i][j][r] + bv);
      }
    }
  }
}

// ---------------------------------------------------------------------------
// Fused attention per (b,h, 128-row sq block):
//  pass1: online row max/sum of masked scores (no writes)
//  pass2: recompute scores, write normalized alignment (f32, once),
//         accumulate context O += P @ V via WMMA, write ctx f16.
// Block: 256 threads / 8 waves. Score tile 128x128 (waves 4x2, 2x4 WMMA);
// PV: wave = 16 rows x 64 cols (4 WMMA accums), k = sk tile (4 steps of 32).
// ---------------------------------------------------------------------------
__global__ void __launch_bounds__(256)
fused_attn_kernel(const _Float16* __restrict__ Qp, const _Float16* __restrict__ Kp,
                  const _Float16* __restrict__ Vp, const float* __restrict__ mask,
                  float* __restrict__ align, _Float16* __restrict__ ctx) {
  __shared__ _Float16 Qs[128 * QSTR];     // Q rows (staged once)
  __shared__ _Float16 Ks[128 * QSTR];     // K rows for current sk tile
  __shared__ _Float16 Vs[64 * PSTR];      // V tile transposed [n][sk_local]
  __shared__ _Float16 Ps[128 * PSTR];     // P tile [sq_local][sk_local]
  __shared__ float m_row[128];            // running row max
  __shared__ float l_row[128];            // running row sum -> 1/sum
  __shared__ float scale_row[128];        // exp(m_old - m_new)
  __shared__ float part[2][128];          // per-wave-column partials

  const int t = threadIdx.x, lane = t & 31, wid = t >> 5;
  const int hsel = lane >> 4, lr = lane & 15;
  const int bh = blockIdx.z;
  const int b = bh / Hc, h = bh % Hc;
  const int sq0 = blockIdx.x * 128;
  const int wm = (wid >> 1) * 32;
  const int wn = (wid & 1) * 64;

  const _Float16* Qh = Qp + (size_t)b * Sc * Dc + h * Kc;
  const _Float16* Kh = Kp + (size_t)b * Sc * Dc + h * Kc;
  const _Float16* Vh = Vp + (size_t)b * Sc * Dc + h * Kc;
  const float*    mp = mask + (size_t)b * Sc * Sc;       // [sq][sk]
  float*          ab = align + (size_t)bh * Sc * Sc;     // [sq][sk]

  // stage Q block (128 x 64 f16) once
  {
    const int r = t >> 1, c = (t & 1) * 32;
    const v8h* s = (const v8h*)(Qh + (size_t)(sq0 + r) * Dc + c);
    v8h* d = (v8h*)(Qs + r * QSTR + c);
    d[0] = s[0]; d[1] = s[1]; d[2] = s[2]; d[3] = s[3];
  }
  if (t < 128) { m_row[t] = -3.0e38f; l_row[t] = 0.0f; }

  // ---------------- pass 1: row statistics ----------------
  for (int skt = 0; skt < Sc / 128; ++skt) {
    const int sk0 = skt * 128;
    __syncthreads();                      // Ks reuse + Q staging on first iter
    {
      const int r = t >> 1, c = (t & 1) * 32;
      const v8h* s = (const v8h*)(Kh + (size_t)(sk0 + r) * Dc + c);
      v8h* d = (v8h*)(Ks + r * QSTR + c);
      d[0] = s[0]; d[1] = s[1]; d[2] = s[2]; d[3] = s[3];
    }
    __syncthreads();

    v8f acc[2][4] = {};
#pragma unroll
    for (int kb = 0; kb < Kc; kb += 32) {
      v16h af[2], bf[4];
#pragma unroll
      for (int i = 0; i < 2; ++i) af[i] = load_frag_s(Qs, QSTR, wm + i * 16, kb, lane);
#pragma unroll
      for (int j = 0; j < 4; ++j) bf[j] = load_frag_s(Ks, QSTR, wn + j * 16, kb, lane);
#pragma unroll
      for (int i = 0; i < 2; ++i)
#pragma unroll
        for (int j = 0; j < 4; ++j)
          acc[i][j] = wmma16x16x32(af[i], bf[j], acc[i][j]);
    }

    // scale + mask (exact zeros -> -1e9), in place
#pragma unroll
    for (int i = 0; i < 2; ++i)
#pragma unroll
      for (int r = 0; r < 8; ++r) {
        const int sq = sq0 + wm + i * 16 + hsel * 8 + r;
#pragma unroll
        for (int j = 0; j < 4; ++j) {
          const int sk = sk0 + wn + j * 16 + lr;
          float x = acc[i][j][r] * 0.125f * mp[(size_t)sq * Sc + sk];
          acc[i][j][r] = (x == 0.0f) ? -1.0e9f : x;
        }
      }

    // tile row max: in-lane over j, shuffle over 16-lane column group
#pragma unroll
    for (int i = 0; i < 2; ++i)
#pragma unroll
      for (int r = 0; r < 8; ++r) {
        float pm = acc[i][0][r];
#pragma unroll
        for (int j = 1; j < 4; ++j) pm = fmaxf(pm, acc[i][j][r]);
        pm = fmaxf(pm, __shfl_xor(pm, 1, 32));
        pm = fmaxf(pm, __shfl_xor(pm, 2, 32));
        pm = fmaxf(pm, __shfl_xor(pm, 4, 32));
        pm = fmaxf(pm, __shfl_xor(pm, 8, 32));
        if (lr == 0) part[wid & 1][wm + i * 16 + hsel * 8 + r] = pm;
      }
    __syncthreads();
    if (t < 128) {
      const float tm = fmaxf(part[0][t], part[1][t]);
      const float mo = m_row[t];
      const float mn = fmaxf(mo, tm);
      m_row[t] = mn;
      scale_row[t] = __expf(mo - mn);
    }
    __syncthreads();

    // tile row sum of exp(x - m_new)
#pragma unroll
    for (int i = 0; i < 2; ++i)
#pragma unroll
      for (int r = 0; r < 8; ++r) {
        const int row = wm + i * 16 + hsel * 8 + r;
        const float mrow = m_row[row];
        float ps = 0.0f;
#pragma unroll
        for (int j = 0; j < 4; ++j) ps += __expf(acc[i][j][r] - mrow);
        ps += __shfl_xor(ps, 1, 32);
        ps += __shfl_xor(ps, 2, 32);
        ps += __shfl_xor(ps, 4, 32);
        ps += __shfl_xor(ps, 8, 32);
        if (lr == 0) part[wid & 1][row] = ps;
      }
    __syncthreads();
    if (t < 128)
      l_row[t] = l_row[t] * scale_row[t] + part[0][t] + part[1][t];
  }

  __syncthreads();
  if (t < 128) l_row[t] = 1.0f / l_row[t];   // now inv_l

  // ---------------- pass 2: emit alignment + accumulate O ----------------
  v8f oacc[4] = {};
  for (int skt = 0; skt < Sc / 128; ++skt) {
    const int sk0 = skt * 128;
    __syncthreads();                      // Ks/Vs/Ps reuse (and inv_l on first iter)
    {
      const int r = t >> 1, c = (t & 1) * 32;
      const v8h* s = (const v8h*)(Kh + (size_t)(sk0 + r) * Dc + c);
      v8h* d = (v8h*)(Ks + r * QSTR + c);
      d[0] = s[0]; d[1] = s[1]; d[2] = s[2]; d[3] = s[3];
    }
    {
      const int r = t >> 1, c = (t & 1) * 32;
      const v8h* s = (const v8h*)(Vh + (size_t)(sk0 + r) * Dc + c);
#pragma unroll
      for (int q4 = 0; q4 < 4; ++q4) {
        v8h vv = s[q4];
#pragma unroll
        for (int e = 0; e < 8; ++e) Vs[(c + q4 * 8 + e) * PSTR + r] = vv[e];
      }
    }
    __syncthreads();

    v8f acc[2][4] = {};
#pragma unroll
    for (int kb = 0; kb < Kc; kb += 32) {
      v16h af[2], bf[4];
#pragma unroll
      for (int i = 0; i < 2; ++i) af[i] = load_frag_s(Qs, QSTR, wm + i * 16, kb, lane);
#pragma unroll
      for (int j = 0; j < 4; ++j) bf[j] = load_frag_s(Ks, QSTR, wn + j * 16, kb, lane);
#pragma unroll
      for (int i = 0; i < 2; ++i)
#pragma unroll
        for (int j = 0; j < 4; ++j)
          acc[i][j] = wmma16x16x32(af[i], bf[j], acc[i][j]);
    }

    // p = exp(x - m)/l ; write alignment (global, f32) and Ps (LDS, f16)
#pragma unroll
    for (int i = 0; i < 2; ++i)
#pragma unroll
      for (int r = 0; r < 8; ++r) {
        const int row = wm + i * 16 + hsel * 8 + r;
        const int sq = sq0 + row;
        const float mrow = m_row[row];
        const float il = l_row[row];
#pragma unroll
        for (int j = 0; j < 4; ++j) {
          const int skl = wn + j * 16 + lr;
          const int sk = sk0 + skl;
          float x = acc[i][j][r] * 0.125f * mp[(size_t)sq * Sc + sk];
          x = (x == 0.0f) ? -1.0e9f : x;
          const float p = __expf(x - mrow) * il;
          ab[(size_t)sq * Sc + sk] = p;
          Ps[row * PSTR + skl] = (_Float16)p;
        }
      }
    __syncthreads();

    // O += P @ V  (K dim = 128 sk, 4 steps)
#pragma unroll
    for (int kk = 0; kk < 128; kk += 32) {
      v16h af = load_frag_s(Ps, PSTR, wid * 16, kk, lane);
      v16h bf[4];
#pragma unroll
      for (int j = 0; j < 4; ++j) bf[j] = load_frag_s(Vs, PSTR, j * 16, kk, lane);
#pragma unroll
      for (int j = 0; j < 4; ++j) oacc[j] = wmma16x16x32(af, bf[j], oacc[j]);
    }
  }

  // context out, head-interleaved f16: ctx[b*S + s][h*64 + col]
#pragma unroll
  for (int j = 0; j < 4; ++j) {
    const int col = h * Kc + j * 16 + lr;
#pragma unroll
    for (int r = 0; r < 8; ++r) {
      const int s = sq0 + wid * 16 + hsel * 8 + r;
      ctx[((size_t)b * Sc + s) * Dc + col] = (_Float16)oacc[j][r];
    }
  }
}

// ---------------------------------------------------------------------------
extern "C" void kernel_launch(void* const* d_in, const int* in_sizes, int n_in,
                              void* d_out, int out_size, void* d_ws, size_t ws_size,
                              hipStream_t stream) {
  (void)in_sizes; (void)n_in; (void)out_size; (void)ws_size;

  const float* q    = (const float*)d_in[0];
  const float* v    = (const float*)d_in[1];
  const float* mask = (const float*)d_in[2];
  const float* wq_w = (const float*)d_in[3];
  const float* wq_b = (const float*)d_in[4];
  const float* wk_w = (const float*)d_in[5];
  const float* wk_b = (const float*)d_in[6];
  const float* wv_w = (const float*)d_in[7];
  const float* wv_b = (const float*)d_in[8];
  const float* wo_w = (const float*)d_in[9];
  const float* wo_b = (const float*)d_in[10];

  const size_t BSD = (size_t)Bc * Sc * Dc;
  float* heads = (float*)d_out;
  float* align = heads + BSD;

  _Float16* Qp  = (_Float16*)d_ws;
  _Float16* Kp  = Qp + BSD;
  _Float16* Vp  = Kp + BSD;
  _Float16* ctx = Vp + BSD;

  const int M = Bc * Sc;                         // 4096
  const dim3 blk(256);
  const dim3 gproj(M / 128, Dc / 128);           // (32, 8)

  // 1) Q/K/V projections: f32 in, f16 out
  gemm_bias_kernel<float, _Float16><<<gproj, blk, 0, stream>>>(q, wq_w, wq_b, Qp, M, Dc, Dc);
  gemm_bias_kernel<float, _Float16><<<gproj, blk, 0, stream>>>(v, wk_w, wk_b, Kp, M, Dc, Dc);
  gemm_bias_kernel<float, _Float16><<<gproj, blk, 0, stream>>>(v, wv_w, wv_b, Vp, M, Dc, Dc);

  // 2) fused score + masked softmax + P@V; writes alignment once
  fused_attn_kernel<<<dim3(Sc / 128, 1, Bc * Hc), blk, 0, stream>>>(
      Qp, Kp, Vp, mask, align, ctx);

  // 3) output projection: f16 in, f32 out -> heads
  gemm_bias_kernel<_Float16, float><<<gproj, blk, 0, stream>>>(ctx, wo_w, wo_b, heads, M, Dc, Dc);
}
